// Attention_32031866094203
// MI455X (gfx1250) — matrix operand order repaired and checked by reading, not compiled, not run on hardware
//
#include <hip/hip_runtime.h>
#include <hip/hip_bf16.h>
#include <math.h>

typedef _Float16 h16;
typedef __attribute__((ext_vector_type(16))) _Float16 v16h;
typedef __attribute__((ext_vector_type(8)))  _Float16 v8h;
typedef __attribute__((ext_vector_type(8)))  float    v8f;

#define N_TOK   8192
#define DIM     1024
#define DH      64
#define DIM_OUT 1024
#define QKV_COLS 192
#define MASK_VAL (-1e10f)

static __device__ inline v8f wmma_f16(v16h a, v16h b, v8f c) {
  // 8 args: (neg_a, A, neg_b, B, c_mod, C, reuse_a, reuse_b)
  return __builtin_amdgcn_wmma_f32_16x16x32_f16(false, a, false, b, (short)0, c,
                                                false, false);
}

// Async DMA: 16 B per lane, global -> LDS, tracked by ASYNCcnt (no VGPR data).
// LDS byte offset = low 32 bits of the flat shared-memory address.
static __device__ inline void async_ld_b128(h16* lds_dst, const h16* gsrc) {
  const unsigned int lds_off = (unsigned int)(unsigned long long)lds_dst;
  asm volatile("global_load_async_to_lds_b128 %0, %1, off"
               :
               : "v"(lds_off), "v"(gsrc)
               : "memory");
}

// ---- fragment loaders (layouts per CDNA5 ISA 7.12.2) --------------------

// A-frag 16x32 (MxK) from f32 row-major [., stride]; row m, K base kb.
// lane<16: halves -> K = kb + {0..7, 16..23}; lane>=16: K = kb + {8..15, 24..31}
static __device__ inline v16h a_frag_f32(const float* __restrict__ p, int stride,
                                         int m, int kb, int lane) {
  const int sel = (lane & 16) ? 8 : 0;
  const float* r = p + (size_t)m * stride + kb + sel;
  v16h a;
#pragma unroll
  for (int h = 0; h < 8; ++h) a[h] = (h16)r[h];
#pragma unroll
  for (int h = 0; h < 8; ++h) a[8 + h] = (h16)r[16 + h];
  return a;
}

// Same but f16 source, with scale folded in.
static __device__ inline v16h a_frag_f16(const h16* __restrict__ p, int stride,
                                         int m, int kb, int lane, h16 scale) {
  const int sel = (lane & 16) ? 8 : 0;
  const h16* r = p + (size_t)m * stride + kb + sel;
  v16h a;
#pragma unroll
  for (int h = 0; h < 8; ++h) a[h] = r[h] * scale;
#pragma unroll
  for (int h = 0; h < 8; ++h) a[8 + h] = r[16 + h] * scale;
  return a;
}

// B-frag 32x16 (KxN) from a K-contiguous f16 matrix: element B[k][n] stored at
// p[n*stride + k]. Per-lane reads: lane<16 K = kb + {0..7, 16..23}; lane>=16 K = kb + {8..15, 24..31}.
// Works for global or LDS pointers.
static __device__ inline v16h b_frag_T16(const h16* p, int stride, int n,
                                         int kb, int lane) {
  const int bse = (lane & 16) ? 8 : 0;
  const h16* r = p + (size_t)n * stride + kb + bse;
  v16h b;
#pragma unroll
  for (int h = 0; h < 8; ++h) b[h] = r[h];
#pragma unroll
  for (int h = 0; h < 8; ++h) b[8 + h] = r[16 + h];
  return b;
}

// ---- kernel 0: one-time weight transpose + f32->f16 convert -------------
__global__ __launch_bounds__(256) void prep_weights(
    const float* __restrict__ Wqkv, const float* __restrict__ Wout,
    h16* __restrict__ WqkvT, h16* __restrict__ WoutT) {
  const int tid = blockIdx.x * 256 + threadIdx.x;
  if (tid < QKV_COLS * DIM) {
    const int n = tid / DIM, kx = tid % DIM;
    WqkvT[tid] = (h16)Wqkv[(size_t)kx * QKV_COLS + n];
  } else {
    const int t2 = tid - QKV_COLS * DIM;  // < DIM_OUT * DH
    const int n = t2 / DH, kx = t2 % DH;
    WoutT[t2] = (h16)Wout[(size_t)kx * DIM_OUT + n];
  }
}

// ---- kernel 1: qkv = x @ Wqkv; write q,k row-major f16, v transposed ----
__global__ __launch_bounds__(128) void qkv_gemm(const float* __restrict__ x,
                                                const h16* __restrict__ WqkvT,
                                                h16* __restrict__ q,
                                                h16* __restrict__ k,
                                                h16* __restrict__ vT) {
  const int lane  = threadIdx.x & 31;
  const int wave  = threadIdx.x >> 5;
  const int mbase = blockIdx.x * 16;
  const int m     = mbase + (lane & 15);
  const v8f zacc = {0.f, 0.f, 0.f, 0.f, 0.f, 0.f, 0.f, 0.f};
  v8f acc[3] = {zacc, zacc, zacc};

  for (int kb = 0; kb < DIM; kb += 32) {
    const v16h a = a_frag_f32(x, DIM, m, kb, lane);
#pragma unroll
    for (int t = 0; t < 3; ++t) {
      const int n = wave * 48 + t * 16 + (lane & 15);
      const v16h b = b_frag_T16(WqkvT, DIM, n, kb, lane);
      acc[t] = wmma_f16(a, b, acc[t]);
    }
  }

  const int rsel = (lane & 16) ? 8 : 0;
#pragma unroll
  for (int t = 0; t < 3; ++t) {
    const int col = wave * 48 + t * 16 + (lane & 15);
    if (col < 128) {
      h16* dst = (col < 64) ? q : k;
      const int c = col & 63;
#pragma unroll
      for (int r = 0; r < 8; ++r)
        dst[(size_t)(mbase + rsel + r) * DH + c] = (h16)acc[t][r];
    } else {
      // v stored transposed: vT[d][token]; rows contiguous -> one 16B store
      const int c = col - 128;
      v8h pk;
#pragma unroll
      for (int r = 0; r < 8; ++r) pk[r] = (h16)acc[t][r];
      *(v8h*)(vT + (size_t)c * N_TOK + mbase + rsel) = pk;
    }
  }
}

// ---- kernel 2: causal flash attention, o = softmax(q k^T / 8) v ---------
// grid: 128 blocks x 4 waves; wave owns 16 query rows. K/V tiles (32 keys)
// are double-buffered in LDS via GLOBAL_LOAD_ASYNC_TO_LDS_B128 (4 async ops
// per wave per tile -> s_wait_asynccnt 4 overlaps DMA with WMMA+softmax).
__global__ __launch_bounds__(128) void flash_attn(const h16* __restrict__ q,
                                                  const h16* __restrict__ kk,
                                                  const h16* __restrict__ vT,
                                                  h16* __restrict__ o) {
  __shared__ __align__(16) h16 kbuf[2][32 * DH];  // [key][d]   4 KB each
  __shared__ __align__(16) h16 vbuf[2][DH * 32];  // [d][key]   4 KB each
  __shared__ h16 lds_p[4][16 * 32];

  const int tid   = threadIdx.x;
  const int lane  = tid & 31;
  const int wave  = tid >> 5;
  const int rbb   = blockIdx.x * 64;       // block's first row
  const int rbase = rbb + wave * 16;       // wave's first row
  const int mrow  = rbase + (lane & 15);
  const int rsel  = (lane & 16) ? 8 : 0;
  const v8f zacc = {0.f, 0.f, 0.f, 0.f, 0.f, 0.f, 0.f, 0.f};

  // q rows, with 1/sqrt(64)=0.125 folded into A (exact in f16)
  const v16h aq0 = a_frag_f16(q, DH, mrow, 0, lane, (h16)0.125f);
  const v16h aq1 = a_frag_f16(q, DH, mrow, 32, lane, (h16)0.125f);

  v8f Oacc[4] = {zacc, zacc, zacc, zacc};
  float m_s[8], l_s[8];
#pragma unroll
  for (int r = 0; r < 8; ++r) { m_s[r] = -1e30f; l_s[r] = 0.f; }

  const int njt = (rbb + 63) / 32 + 1;  // block-uniform causal bound

  // stage tile 0 into buffer 0 (each thread moves 4 x 16 B)
#pragma unroll
  for (int i = 0; i < 2; ++i) {
    const int c = tid + i * 128;
    async_ld_b128(&kbuf[0][(c >> 3) * DH + (c & 7) * 8],
                  kk + (size_t)(c >> 3) * DH + (c & 7) * 8);
    async_ld_b128(&vbuf[0][(c >> 2) * 32 + (c & 3) * 8],
                  vT + (size_t)(c >> 2) * N_TOK + (c & 3) * 8);
  }

  for (int jt = 0; jt < njt; ++jt) {
    const int jbase = jt * 32;
    const int buf = jt & 1;

    // kick off DMA for the next tile, then wait for this tile's DMA
    if (jt + 1 < njt) {
      const int nj = jbase + 32;
      const int nb = buf ^ 1;
#pragma unroll
      for (int i = 0; i < 2; ++i) {
        const int c = tid + i * 128;
        async_ld_b128(&kbuf[nb][(c >> 3) * DH + (c & 7) * 8],
                      kk + (size_t)(nj + (c >> 3)) * DH + (c & 7) * 8);
        async_ld_b128(&vbuf[nb][(c >> 2) * 32 + (c & 3) * 8],
                      vT + (size_t)(c >> 2) * N_TOK + nj + (c & 3) * 8);
      }
      asm volatile("s_wait_asynccnt 0x4" ::: "memory");
    } else {
      asm volatile("s_wait_asynccnt 0x0" ::: "memory");
    }
    __syncthreads();  // tile jt visible to all waves

    if (jbase <= rbase + 15) {  // wave has unmasked keys in this tile
      const h16* kb_c = &kbuf[buf][0];
      const h16* vb_c = &vbuf[buf][0];

      // S tile 16x32 = two 16x16 f32 frags, K = dh = 64
      const v16h bk00 = b_frag_T16(kb_c, DH, (lane & 15), 0, lane);
      const v16h bk01 = b_frag_T16(kb_c, DH, (lane & 15), 32, lane);
      const v16h bk10 = b_frag_T16(kb_c, DH, 16 + (lane & 15), 0, lane);
      const v16h bk11 = b_frag_T16(kb_c, DH, 16 + (lane & 15), 32, lane);
      v8f S0 = zacc, S1 = zacc;
      S0 = wmma_f16(aq0, bk00, S0);
      S0 = wmma_f16(aq1, bk01, S0);
      S1 = wmma_f16(aq0, bk10, S1);
      S1 = wmma_f16(aq1, bk11, S1);

      // causal mask (only the diagonal band needs it)
      if (jbase + 31 > rbase) {
        const int j0 = jbase + (lane & 15);
        const int j1 = j0 + 16;
#pragma unroll
        for (int r = 0; r < 8; ++r) {
          const int row = rbase + rsel + r;
          if (j0 > row) S0[r] = MASK_VAL;
          if (j1 > row) S1[r] = MASK_VAL;
        }
      }

      // online softmax: row stats via cross-lane (width-16) reductions
#pragma unroll
      for (int r = 0; r < 8; ++r) {
        float mx = fmaxf(S0[r], S1[r]);
#pragma unroll
        for (int off = 1; off < 16; off <<= 1)
          mx = fmaxf(mx, __shfl_xor(mx, off, 16));
        const float mn = fmaxf(m_s[r], mx);
        const float p0 = __expf(S0[r] - mn);
        const float p1 = __expf(S1[r] - mn);
        float rs = p0 + p1;
#pragma unroll
        for (int off = 1; off < 16; off <<= 1) rs += __shfl_xor(rs, off, 16);
        const float alpha = __expf(m_s[r] - mn);
        l_s[r] = l_s[r] * alpha + rs;
        m_s[r] = mn;
        Oacc[0][r] *= alpha; Oacc[1][r] *= alpha;
        Oacc[2][r] *= alpha; Oacc[3][r] *= alpha;
        S0[r] = p0; S1[r] = p1;
      }

      // P: C-layout (col-in-lane) -> A-frag layout via wave-private LDS
      h16* pl = &lds_p[wave][0];
#pragma unroll
      for (int r = 0; r < 8; ++r) {
        const int row = rsel + r;
        pl[row * 32 + (lane & 15)]      = (h16)S0[r];
        pl[row * 32 + 16 + (lane & 15)] = (h16)S1[r];
      }
      v16h aP;
      {
        const int sel = (lane & 16) ? 8 : 0;
        const h16* rp = pl + (lane & 15) * 32 + sel;
#pragma unroll
        for (int h = 0; h < 8; ++h) aP[h] = rp[h];
#pragma unroll
        for (int h = 0; h < 8; ++h) aP[8 + h] = rp[16 + h];
      }

      // O += P @ V_tile  (four 16-col d tiles; vbuf is K-contiguous per lane)
#pragma unroll
      for (int dt = 0; dt < 4; ++dt) {
        const v16h bv = b_frag_T16(vb_c, 32, dt * 16 + (lane & 15), 0, lane);
        Oacc[dt] = wmma_f16(aP, bv, Oacc[dt]);
      }
    }

    __syncthreads();  // all waves done with buf before it is re-filled
  }

  // normalize and store o (f16) for the output projection
#pragma unroll
  for (int dt = 0; dt < 4; ++dt)
#pragma unroll
    for (int r = 0; r < 8; ++r) {
      const int row = rbase + rsel + r;
      o[(size_t)row * DH + dt * 16 + (lane & 15)] =
          (h16)(Oacc[dt][r] / l_s[r]);
    }
}

// ---- kernel 3: out = o @ Wout + bout ------------------------------------
__global__ __launch_bounds__(256) void out_gemm(const h16* __restrict__ o,
                                                const h16* __restrict__ WoutT,
                                                const float* __restrict__ bout,
                                                float* __restrict__ out) {
  const int lane  = threadIdx.x & 31;
  const int wave  = threadIdx.x >> 5;
  const int mbase = blockIdx.x * 16;
  const int m     = mbase + (lane & 15);
  const int rsel  = (lane & 16) ? 8 : 0;
  const v8f zacc = {0.f, 0.f, 0.f, 0.f, 0.f, 0.f, 0.f, 0.f};

  const v16h a0 = a_frag_f16(o, DH, m, 0, lane, (h16)1.0f);
  const v16h a1 = a_frag_f16(o, DH, m, 32, lane, (h16)1.0f);

#pragma unroll
  for (int t = 0; t < 8; ++t) {
    const int n = wave * 128 + t * 16 + (lane & 15);
    const v16h b0 = b_frag_T16(WoutT, DH, n, 0, lane);
    const v16h b1 = b_frag_T16(WoutT, DH, n, 32, lane);
    v8f acc = zacc;
    acc = wmma_f16(a0, b0, acc);
    acc = wmma_f16(a1, b1, acc);
    const float bias = bout[n];
#pragma unroll
    for (int r = 0; r < 8; ++r)
      out[(size_t)(mbase + rsel + r) * DIM_OUT + n] = acc[r] + bias;
  }
}

extern "C" void kernel_launch(void* const* d_in, const int* in_sizes, int n_in,
                              void* d_out, int out_size, void* d_ws,
                              size_t ws_size, hipStream_t stream) {
  const float* x    = (const float*)d_in[0];
  const float* Wqkv = (const float*)d_in[1];
  const float* Wout = (const float*)d_in[2];
  const float* bout = (const float*)d_in[3];
  float* out = (float*)d_out;

  // workspace: q | k | vT | o  (each N_TOK*DH f16 = 1 MB)
  //            | WqkvT (192*1024 f16 = 384 KB) | WoutT (1024*64 f16 = 128 KB)
  char* ws = (char*)d_ws;
  h16* q     = (h16*)(ws);
  h16* k     = (h16*)(ws + (size_t)N_TOK * DH * 2);
  h16* vT    = (h16*)(ws + (size_t)N_TOK * DH * 4);
  h16* o     = (h16*)(ws + (size_t)N_TOK * DH * 6);
  h16* WqkvT = (h16*)(ws + (size_t)N_TOK * DH * 8);
  h16* WoutT = (h16*)(ws + (size_t)N_TOK * DH * 8 +
                      (size_t)QKV_COLS * DIM * 2);

  hipLaunchKernelGGL(prep_weights,
                     dim3((QKV_COLS * DIM + DIM_OUT * DH) / 256), dim3(256), 0,
                     stream, Wqkv, Wout, WqkvT, WoutT);
  hipLaunchKernelGGL(qkv_gemm, dim3(N_TOK / 16), dim3(128), 0, stream, x,
                     WqkvT, q, k, vT);
  hipLaunchKernelGGL(flash_attn, dim3(N_TOK / 64), dim3(128), 0, stream, q, k,
                     vT, o);
  hipLaunchKernelGGL(out_gemm, dim3(N_TOK / 16), dim3(256), 0, stream, o,
                     WoutT, bout, out);
}
